// SV_PointNet_PSEG_63007170232738
// MI455X (gfx1250) — compile-verified, matrix operand order checked by reading, not run on hardware
//
#include <hip/hip_runtime.h>
#include <hip/hip_bf16.h>

typedef long long i64;
typedef __bf16 bf16;
typedef __bf16 v16bf __attribute__((ext_vector_type(16)));
typedef float  v8f   __attribute__((ext_vector_type(8)));

#define KNN 20
#define EPSF 1e-6f
#define KU 64   // K-depth per pipeline stage: two v_wmma 16x16x32 per stage

__device__ __forceinline__ bf16 f2bf(float f) {
  unsigned u = __float_as_uint(f);
  unsigned r = (u + 0x7FFFu + ((u >> 16) & 1u)) >> 16;
  unsigned short h = (unsigned short)r;
  return *(bf16*)&h;
}

// ---------------- kNN: brute force top-20 with LDS point tiles ----------------
__global__ __launch_bounds__(256) void knn_k(const float* __restrict__ x, int Nc,
                                             int* __restrict__ idx) {
  int b = blockIdx.y;
  int n = blockIdx.x * 256 + threadIdx.x;
  const float* xb = x + (size_t)b * 3 * Nc;
  float px = xb[n], py = xb[Nc + n], pz = xb[2 * Nc + n];
  float bd[KNN]; int bi[KNN];
#pragma unroll
  for (int k = 0; k < KNN; ++k) { bd[k] = 3.0e38f; bi[k] = 0; }
  __shared__ float sx[256], sy[256], sz[256];
  for (int t0 = 0; t0 < Nc; t0 += 256) {
    int j = t0 + threadIdx.x;
    sx[threadIdx.x] = xb[j]; sy[threadIdx.x] = xb[Nc + j]; sz[threadIdx.x] = xb[2 * Nc + j];
    __syncthreads();
    for (int jj = 0; jj < 256; ++jj) {
      float dx = px - sx[jj], dy = py - sy[jj], dz = pz - sz[jj];
      float d = dx * dx + dy * dy + dz * dz;
      if (d < bd[KNN - 1]) {
        int p = KNN - 1;
        while (p > 0 && bd[p - 1] > d) { bd[p] = bd[p - 1]; bi[p] = bi[p - 1]; --p; }
        bd[p] = d; bi[p] = t0 + jj;
      }
    }
    __syncthreads();
  }
  int* o = idx + ((size_t)b * Nc + n) * KNN;
#pragma unroll
  for (int k = 0; k < KNN; ++k) o[k] = bi[k];
}

// ------- edge features: v=(nbr-ctr, ctr, nbr x ctr); 9 invariants + vnorms ----
__global__ __launch_bounds__(256) void edge_k(const float* __restrict__ x,
    const int* __restrict__ idx, const float* __restrict__ v2s, int Bc, int Nc, int Kn,
    bf16* __restrict__ As, int KpS, bf16* __restrict__ Av, int KpV) {
  i64 i = (i64)blockIdx.x * 256 + threadIdx.x;
  i64 tot = (i64)Bc * Nc * Kn;
  if (i >= tot) return;
  int j = idx[i];
  i64 bn = i / Kn; int n = (int)(bn % Nc); int b = (int)(bn / Nc);
  const float* xb = x + (size_t)b * 3 * Nc;
  float cx = xb[n], cy = xb[Nc + n], cz = xb[2 * Nc + n];
  float nx = xb[j], ny = xb[Nc + j], nz = xb[2 * Nc + j];
  float v[3][3];
  v[0][0] = nx - cx; v[1][0] = ny - cy; v[2][0] = nz - cz;
  v[0][1] = cx;      v[1][1] = cy;      v[2][1] = cz;
  v[0][2] = ny * cz - nz * cy; v[1][2] = nz * cx - nx * cz; v[2][2] = nx * cy - ny * cx;
  float v2[3][3];
#pragma unroll
  for (int d = 0; d < 3; ++d)
#pragma unroll
    for (int e = 0; e < 3; ++e)
      v2[d][e] = v[d][0] * v2s[e] + v[d][1] * v2s[3 + e] + v[d][2] * v2s[6 + e];
  bf16* as = As + (size_t)i * KpS;
#pragma unroll
  for (int c = 0; c < 3; ++c)
#pragma unroll
    for (int e = 0; e < 3; ++e)
      as[c * 3 + e] = f2bf(v[0][c] * v2[0][e] + v[1][c] * v2[1][e] + v[2][c] * v2[2][e]);
#pragma unroll
  for (int c = 0; c < 3; ++c)
    as[9 + c] = f2bf(sqrtf(v[0][c] * v[0][c] + v[1][c] * v[1][c] + v[2][c] * v[2][c] + EPSF));
  for (int k = 12; k < KpS; ++k) as[k] = f2bf(0.f);
#pragma unroll
  for (int d = 0; d < 3; ++d) {
    bf16* av = Av + ((size_t)i * 3 + d) * KpV;
    av[0] = f2bf(v[d][0]); av[1] = f2bf(v[d][1]); av[2] = f2bf(v[d][2]);
    for (int k = 3; k < KpV; ++k) av[k] = f2bf(0.f);
  }
}

// ---------------- WMMA GEMM: C[M,N] = A[M,Kp](bf16) * BT[N,Kp]^T -------------
// Requirements: M % 128 == 0 (holds for every layer here), Kp % 64 == 0.
// 8 waves -> 128x16 tile; double-buffered LDS; 2 x v_wmma_f32_16x16x32_bf16 per
// barrier; global loads for stage i+1 issued before computing stage i.
__global__ __launch_bounds__(256) void gemm_bf16_k(const bf16* __restrict__ A,
    const bf16* __restrict__ BT, float* __restrict__ C, int M, int Ncols, int Kp,
    int ldc, int coff) {
  __shared__ bf16 lA[2][128 * KU];   // 2 x 16 KB
  __shared__ bf16 lB[2][16 * KU];    // 2 x  2 KB
  int t = threadIdx.x;
  int wave = t >> 5, lane = t & 31;
  int mtile = blockIdx.x * 128, ntile = blockIdx.y * 16;
  int m16 = lane & 15, half = lane >> 4;
  bool bldr = (t < 64);
  union { v8f v; float f[8]; } acc;
#pragma unroll
  for (int r = 0; r < 8; ++r) acc.f[r] = 0.f;

  uint4 ra[4], rb[2];
  auto gload = [&](int k0) {
#pragma unroll
    for (int i = 0; i < 4; ++i) {            // A stage: 128x64 bf16 = 1024 uint4
      int q = t + i * 256;
      int row = q >> 3, qc = q & 7;
      ra[i] = *(const uint4*)(A + (size_t)(mtile + row) * Kp + k0 + qc * 8);
    }
    if (bldr) {
#pragma unroll
      for (int i = 0; i < 2; ++i) {          // B stage: 16x64 bf16 = 128 uint4
        int q = t + i * 64;
        int row = q >> 3, qc = q & 7;
        uint4 z = make_uint4(0u, 0u, 0u, 0u);
        rb[i] = (ntile + row < Ncols)
                  ? *(const uint4*)(BT + (size_t)(ntile + row) * Kp + k0 + qc * 8) : z;
      }
    }
  };
  auto lstore = [&](int buf) {
#pragma unroll
    for (int i = 0; i < 4; ++i) {
      int q = t + i * 256;
      int row = q >> 3, qc = q & 7;
      *(uint4*)(&lA[buf][row * KU + qc * 8]) = ra[i];
    }
    if (bldr) {
#pragma unroll
      for (int i = 0; i < 2; ++i) {
        int q = t + i * 64;
        int row = q >> 3, qc = q & 7;
        *(uint4*)(&lB[buf][row * KU + qc * 8]) = rb[i];
      }
    }
  };
  auto compute = [&](int buf) {
#pragma unroll
    for (int s = 0; s < 2; ++s) {
      union { v16bf v; uint4 q[2]; } af, bfv;
      // 16-bit A 16x32 layout: elems 0..7 -> K=half*8+e, elems 8..15 -> K=16+half*8+e
      const bf16* pa = &lA[buf][(wave * 16 + m16) * KU + s * 32];
      af.q[0] = *(const uint4*)(pa + half * 8);
      af.q[1] = *(const uint4*)(pa + 16 + half * 8);
      const bf16* pb = &lB[buf][m16 * KU + s * 32];
      bfv.q[0] = *(const uint4*)(pb + half * 8);
      bfv.q[1] = *(const uint4*)(pb + 16 + half * 8);
      acc.v = __builtin_amdgcn_wmma_f32_16x16x32_bf16(false, af.v, false, bfv.v,
                                                      (short)0, acc.v, false, false);
    }
  };

  gload(0);
  lstore(0);
  __syncthreads();
  int nsteps = Kp / KU;
  for (int it = 0; it < nsteps; ++it) {
    int cur = it & 1;
    if (it + 1 < nsteps) {
      gload((it + 1) * KU);
      // speculative prefetch two stages ahead (dropped silently if past end)
      __builtin_prefetch(A + (size_t)(mtile + (t >> 1)) * Kp + (size_t)(it + 2) * KU
                           + (t & 1) * 32, 0, 0);
    }
    compute(cur);
    if (it + 1 < nsteps) lstore(cur ^ 1);
    __syncthreads();
  }

  int gn = ntile + m16;
  if (gn < Ncols) {
#pragma unroll
    for (int r = 0; r < 8; ++r) {            // C layout: VGPR r -> M = half*8 + r
      int gm = mtile + wave * 16 + half * 8 + r;
      C[(size_t)gm * ldc + coff + gn] = acc.f[r];
    }
  }
}

// ---------------- weight pack: W[K,N] f32 -> WT[N,Kp] bf16 (zero-pad K) ------
__global__ void wpackT_k(const float* __restrict__ W, int K, int Ncols,
                         bf16* __restrict__ WT, int Kp) {
  i64 i = (i64)blockIdx.x * 256 + threadIdx.x;
  if (i >= (i64)Ncols * Kp) return;
  int k = (int)(i % Kp); int n = (int)(i / Kp);
  WT[i] = (k < K) ? f2bf(W[(size_t)k * Ncols + n]) : f2bf(0.f);
}

// ---------------- activation pack: strided f32 -> dense bf16 (zero-pad) ------
__global__ void pack_bf16_k(const float* __restrict__ X, int ldx, int xoff, int M,
                            int C, bf16* __restrict__ Y, int Kp) {
  i64 i = (i64)blockIdx.x * 256 + threadIdx.x;
  if (i >= (i64)M * Kp) return;
  int k = (int)(i % Kp); i64 m = i / Kp;
  Y[i] = (k < C) ? f2bf(X[(size_t)m * ldx + xoff + k]) : f2bf(0.f);
}

__global__ void copy2d_k(const float* __restrict__ X, int ldx, int xoff, int M, int C,
                         float* __restrict__ Y, int ldy, int yoff) {
  i64 i = (i64)blockIdx.x * 256 + threadIdx.x;
  if (i >= (i64)M * C) return;
  int c = (int)(i % C); i64 m = i / C;
  Y[(size_t)m * ldy + yoff + c] = X[(size_t)m * ldx + xoff + c];
}

// ---------------- vnorm of (M,3,C) vectors into scalar concat ----------------
__global__ void vnorm_k(const float* __restrict__ V, int ldv, int voff, int M, int C,
                        float* __restrict__ S, int lds_, int soff) {
  i64 i = (i64)blockIdx.x * 256 + threadIdx.x;
  if (i >= (i64)M * C) return;
  int c = (int)(i % C); i64 m = i / C;
  const float* p = V + (size_t)m * 3 * ldv + voff + c;
  float a = p[0], b = p[ldv], d = p[2 * ldv];
  S[(size_t)m * lds_ + soff + c] = sqrtf(a * a + b * b + d * d + EPSF);
}

// ---------------- BN (training mode): per-channel mean/var -------------------
__global__ __launch_bounds__(256) void bn_stats_k(const float* __restrict__ X, int M,
                                                  int C, float* mean, float* var) {
  int c = blockIdx.x;
  float s = 0.f, s2 = 0.f;
  for (int m = threadIdx.x; m < M; m += 256) {
    float v = X[(size_t)m * C + c];
    s += v; s2 += v * v;
  }
  __shared__ float sh[512];
  sh[threadIdx.x] = s; sh[256 + threadIdx.x] = s2;
  __syncthreads();
  for (int o = 128; o > 0; o >>= 1) {
    if ((int)threadIdx.x < o) {
      sh[threadIdx.x] += sh[threadIdx.x + o];
      sh[256 + threadIdx.x] += sh[256 + threadIdx.x + o];
    }
    __syncthreads();
  }
  if (threadIdx.x == 0) {
    float mu = sh[0] / M; mean[c] = mu; var[c] = sh[256] / M - mu * mu;
  }
}

// Normalize + ReLU; optional f32 strided out and/or bf16 packed (padded) out.
__global__ void bn_relu_k(const float* __restrict__ X, int M, int C, int Wp,
                          const float* __restrict__ mean, const float* __restrict__ var,
                          const float* __restrict__ g, const float* __restrict__ beta,
                          float* __restrict__ Yf, int ldy, int yoff, bf16* __restrict__ Yb) {
  i64 i = (i64)blockIdx.x * 256 + threadIdx.x;
  if (i >= (i64)M * Wp) return;
  int c = (int)(i % Wp); i64 m = i / Wp;
  if (c >= C) { if (Yb) Yb[(size_t)m * Wp + c] = f2bf(0.f); return; }
  float xv = X[(size_t)m * C + c];
  float y = g[c] * (xv - mean[c]) * rsqrtf(var[c] + 1e-5f) + beta[c];
  y = fmaxf(y, 0.f);
  if (Yf) Yf[(size_t)m * ldy + yoff + c] = y;
  if (Yb) Yb[(size_t)m * Wp + c] = f2bf(y);
}

// ---------------- direction-gated VN-ReLU ------------------------------------
__global__ void vn_gate_k(const float* __restrict__ Q, const float* __restrict__ D,
                          int M, int C, float* __restrict__ VO, int ldv, int voff) {
  i64 i = (i64)blockIdx.x * 256 + threadIdx.x;
  if (i >= (i64)M * C) return;
  int c = (int)(i % C); i64 m = i / C;
  const float* q = Q + (size_t)m * 3 * C + c;
  const float* d = D + (size_t)m * 3 * C + c;
  float q0 = q[0], q1 = q[C], q2 = q[2 * C];
  float d0 = d[0], d1 = d[C], d2 = d[2 * C];
  float dot = q0 * d0 + q1 * d1 + q2 * d2;
  float t = (dot >= 0.f) ? 0.f : dot / (d0 * d0 + d1 * d1 + d2 * d2 + EPSF);
  float* o = VO + (size_t)m * 3 * ldv + voff + c;
  o[0] = q0 - t * d0; o[ldv] = q1 - t * d1; o[2 * ldv] = q2 - t * d2;
}

// ---------------- pools ------------------------------------------------------
__global__ void pool_maxK_k(const float* __restrict__ S, int BNr, int Kn, int C,
                            float* __restrict__ O) {
  i64 i = (i64)blockIdx.x * 256 + threadIdx.x;
  if (i >= (i64)BNr * C) return;
  int c = (int)(i % C); i64 m = i / C;
  float v = -3.0e38f;
  for (int k = 0; k < Kn; ++k) v = fmaxf(v, S[((size_t)m * Kn + k) * C + c]);
  O[(size_t)m * C + c] = v;
}
__global__ void pool_meanK_v_k(const float* __restrict__ V, int BNr, int Kn, int C,
                               float* __restrict__ O) {
  i64 i = (i64)blockIdx.x * 256 + threadIdx.x;
  if (i >= (i64)BNr * 3 * C) return;
  int c = (int)(i % C); i64 t = i / C; int d = (int)(t % 3); i64 m = t / 3;
  float s = 0.f;
  for (int k = 0; k < Kn; ++k) s += V[(((size_t)m * Kn + k) * 3 + d) * C + c];
  O[((size_t)m * 3 + d) * C + c] = s / Kn;
}
__global__ void pool_N_s_k(const float* __restrict__ S, int lds_, int soff, int Bc,
                           int Nc, int C, float* __restrict__ O, int domean) {
  i64 i = (i64)blockIdx.x * 256 + threadIdx.x;
  if (i >= (i64)Bc * C) return;
  int c = (int)(i % C); int b = (int)(i / C);
  float acc = domean ? 0.f : -3.0e38f;
  for (int n = 0; n < Nc; ++n) {
    float v = S[((size_t)b * Nc + n) * lds_ + soff + c];
    acc = domean ? acc + v : fmaxf(acc, v);
  }
  O[(size_t)b * C + c] = domean ? acc / Nc : acc;
}
__global__ void pool_N_v_k(const float* __restrict__ V, int ldv, int voff, int Bc,
                           int Nc, int C, float* __restrict__ O) {
  i64 i = (i64)blockIdx.x * 256 + threadIdx.x;
  if (i >= (i64)Bc * 3 * C) return;
  int c = (int)(i % C); i64 t = i / C; int d = (int)(t % 3); int b = (int)(t / 3);
  float s = 0.f;
  for (int n = 0; n < Nc; ++n) s += V[(((size_t)b * Nc + n) * 3 + d) * ldv + voff + c];
  O[((size_t)b * 3 + d) * C + c] = s / Nc;
}
__global__ void bcast_s_k(const float* __restrict__ G, int Bc, int Nc, int C,
                          float* __restrict__ Y, int ldy, int yoff) {
  i64 i = (i64)blockIdx.x * 256 + threadIdx.x;
  if (i >= (i64)Bc * Nc * C) return;
  int c = (int)(i % C); i64 t = i / C; int n = (int)(t % Nc); int b = (int)(t / Nc);
  Y[((size_t)b * Nc + n) * ldy + yoff + c] = G[(size_t)b * C + c];
}
__global__ void bcast_v_k(const float* __restrict__ G, int Bc, int Nc, int C,
                          float* __restrict__ Y, int ldv, int voff) {
  i64 i = (i64)blockIdx.x * 256 + threadIdx.x;
  if (i >= (i64)Bc * Nc * 3 * C) return;
  int c = (int)(i % C); i64 t = i / C; int d = (int)(t % 3);
  i64 t2 = t / 3; int n = (int)(t2 % Nc); int b = (int)(t2 / Nc);
  Y[(((size_t)b * Nc + n) * 3 + d) * ldv + voff + c] = G[((size_t)b * 3 + d) * C + c];
}

// ---------------- frame projection (vloc) and rotation (cvr) -----------------
__global__ void vloc_k(const float* __restrict__ TR, const float* __restrict__ VF,
                       int M, int C, float* __restrict__ XF, int ldx, int xoff) {
  i64 i = (i64)blockIdx.x * 256 + threadIdx.x;
  if (i >= (i64)M * C) return;
  int c = (int)(i % C); i64 m = i / C;
  const float* tr = TR + (size_t)m * 9;           // tr[d*3+j]
  const float* vf = VF + (size_t)m * 3 * C + c;
  float v0 = vf[0], v1 = vf[C], v2 = vf[2 * C];
  float* o = XF + (size_t)m * ldx + xoff;
#pragma unroll
  for (int j = 0; j < 3; ++j)
    o[j * C + c] = tr[j] * v0 + tr[3 + j] * v1 + tr[6 + j] * v2;
}
__global__ void cvr_k(const float* __restrict__ CV, int ldv, int voff,
                      const float* __restrict__ TR, int M, int C,
                      float* __restrict__ F, int ldf, int foff) {
  i64 i = (i64)blockIdx.x * 256 + threadIdx.x;
  if (i >= (i64)M * C) return;
  int c = (int)(i % C); i64 m = i / C;
  const float* tr = TR + (size_t)m * 9;
  const float* cv = CV + (size_t)m * 3 * ldv + voff + c;
  float v0 = cv[0], v1 = cv[ldv], v2 = cv[2 * ldv];
  float* o = F + (size_t)m * ldf + foff + c * 3;
#pragma unroll
  for (int k = 0; k < 3; ++k)
    o[k] = v0 * tr[k] + v1 * tr[3 + k] + v2 * tr[6 + k];
}

// ---------------- output: bias add + transpose to (B,50,N) -------------------
__global__ void final_k(const float* __restrict__ C50, const float* __restrict__ bias,
                        int Bc, int Nc, float* __restrict__ out) {
  i64 i = (i64)blockIdx.x * 256 + threadIdx.x;
  if (i >= (i64)Bc * 50 * Nc) return;
  int n = (int)(i % Nc); i64 t = i / Nc; int c = (int)(t % 50); int b = (int)(t / 50);
  out[i] = C50[((size_t)b * Nc + n) * 50 + c] + bias[c];
}

#define L1D(tot) <<<dim3((unsigned)((((i64)(tot)) + 255) / 256)), dim3(256), 0, stream>>>

extern "C" void kernel_launch(void* const* d_in, const int* in_sizes, int n_in,
                              void* d_out, int out_size, void* d_ws, size_t ws_size,
                              hipStream_t stream) {
  (void)in_sizes; (void)n_in; (void)out_size; (void)ws_size;
  const int B = 4, N = 2048, BN = B * N, M0 = BN * KNN;
  const float* x   = (const float*)d_in[0];
  const float* lfe = (const float*)d_in[1];
  const float* v2s = (const float*)d_in[2];

  char* wsb = (char*)d_ws; size_t off = 0;
  auto walloc = [&](size_t bytes) -> void* {
    off = (off + 255) & ~(size_t)255;
    void* p = wsb + off; off += bytes; return p;
  };
  auto gemm = [&](const bf16* A, const bf16* WT, float* C, int M, int Ncols, int Kp,
                  int ldc, int coff) {
    dim3 g((M + 127) / 128, (Ncols + 15) / 16);
    gemm_bf16_k<<<g, 256, 0, stream>>>(A, WT, C, M, Ncols, Kp, ldc, coff);
  };
  auto packW = [&](const float* W, int K, int Ncols, int Kp) -> bf16* {
    bf16* WT = (bf16*)walloc((size_t)Ncols * Kp * 2);
    wpackT_k L1D((i64)Ncols * Kp) (W, K, Ncols, WT, Kp);
    return WT;
  };
  // Y = relu(BN(A @ W));  bias-before-BN cancels exactly (mean subtraction)
  auto lin_bn = [&](const bf16* A, int M, int K, int Kp, const float* Wf,
                    const float* g, const float* beta, int Ncols,
                    float* Yf, int ldy, int yoff, bf16* Yb, int Wp) {
    bf16* WT = packW(Wf, K, Ncols, Kp);
    float* C = (float*)walloc((size_t)M * Ncols * 4);
    gemm(A, WT, C, M, Ncols, Kp, Ncols, 0);
    float* mv = (float*)walloc((size_t)2 * Ncols * 4);
    bn_stats_k<<<Ncols, 256, 0, stream>>>(C, M, Ncols, mv, mv + Ncols);
    int W2 = Yb ? Wp : Ncols;
    bn_relu_k L1D((i64)M * W2) (C, M, Ncols, W2, mv, mv + Ncols, g, beta, Yf, ldy, yoff, Yb);
  };
  auto vn_layer = [&](const bf16* V, int M, int K, int Kp, const float* Wv,
                      const float* Wd, int Ncols, float* VO, int ldv, int voff) {
    bf16* WvT = packW(Wv, K, Ncols, Kp);
    bf16* WdT = packW(Wd, K, Ncols, Kp);
    float* Q = (float*)walloc((size_t)M * 3 * Ncols * 4);
    float* D = (float*)walloc((size_t)M * 3 * Ncols * 4);
    gemm(V, WvT, Q, M * 3, Ncols, Kp, Ncols, 0);
    gemm(V, WdT, D, M * 3, Ncols, Kp, Ncols, 0);
    vn_gate_k L1D((i64)M * Ncols) (Q, D, M, Ncols, VO, ldv, voff);
  };
  auto svblock = [&](int pb, const float* Sin, int lds_, int soff, int Cs,
                     const float* Vin, int ldv, int voff, int Cv, int M,
                     int CsO, int CvO, float* SO, int ldso, int sooff,
                     float* VO, int ldvo, int vooff) {
    int Cin = Cs + Cv, Kp = (Cin + 63) & ~63;
    float* SC = (float*)walloc((size_t)M * Cin * 4);
    copy2d_k L1D((i64)M * Cs) (Sin, lds_, soff, M, Cs, SC, Cin, 0);
    vnorm_k  L1D((i64)M * Cv) (Vin, ldv, voff, M, Cv, SC, Cin, Cs);
    bf16* Ab = (bf16*)walloc((size_t)M * Kp * 2);
    pack_bf16_k L1D((i64)M * Kp) (SC, Cin, 0, M, Cin, Ab, Kp);
    lin_bn(Ab, M, Cin, Kp, (const float*)d_in[pb], (const float*)d_in[pb + 2],
           (const float*)d_in[pb + 3], CsO, SO, ldso, sooff, nullptr, 0);
    int Kpv = (Cv + 63) & ~63;
    bf16* Vb = (bf16*)walloc((size_t)M * 3 * Kpv * 2);
    pack_bf16_k L1D((i64)M * 3 * Kpv) (Vin, ldv, voff, M * 3, Cv, Vb, Kpv);
    vn_layer(Vb, M, Cv, Kpv, (const float*)d_in[pb + 4], (const float*)d_in[pb + 5],
             CvO, VO, ldvo, vooff);
  };

  // ---- kNN graph + edge features ----
  int* idx = (int*)walloc((size_t)BN * KNN * 4);
  knn_k<<<dim3(N / 256, B), 256, 0, stream>>>(x, N, idx);
  bf16* As0 = (bf16*)walloc((size_t)M0 * 64 * 2);
  bf16* Av0 = (bf16*)walloc((size_t)M0 * 3 * 64 * 2);
  edge_k L1D((i64)M0) (x, idx, v2s, B, N, KNN, As0, 64, Av0, 64);

  // ---- pos block (params 3..8) + pool over K ----
  float* SC0 = (float*)walloc((size_t)M0 * 32 * 4);
  lin_bn(As0, M0, 12, 64, (const float*)d_in[3], (const float*)d_in[5],
         (const float*)d_in[6], 32, SC0, 32, 0, nullptr, 0);
  float* VO0 = (float*)walloc((size_t)M0 * 3 * 10 * 4);
  vn_layer(Av0, M0, 3, 64, (const float*)d_in[7], (const float*)d_in[8], 10, VO0, 10, 0);
  float* spool = (float*)walloc((size_t)BN * 32 * 4);
  pool_maxK_k   L1D((i64)BN * 32)     (SC0, BN, KNN, 32, spool);
  float* vpool = (float*)walloc((size_t)BN * 3 * 10 * 4);
  pool_meanK_v_k L1D((i64)BN * 3 * 10) (VO0, BN, KNN, 10, vpool);

  // ---- c1..c3 write into wide concat buffers CS(1440)/CV(478) ----
  float* CS = (float*)walloc((size_t)BN * 1440 * 4);
  float* CV = (float*)walloc((size_t)BN * 3 * 478 * 4);
  svblock(9,  spool, 32, 0, 32,  vpool, 10, 0, 10, BN, 32, 10, CS, 1440, 0,   CV, 478, 0);
  svblock(15, CS, 1440, 0, 32,   CV, 478, 0, 10,  BN, 64, 21, CS, 1440, 32,  CV, 478, 10);
  svblock(21, CS, 1440, 32, 64,  CV, 478, 10, 21, BN, 64, 21, CS, 1440, 96,  CV, 478, 31);

  // ---- SV_STNkd: fa, fb, global pool, broadcast ----
  float* SA = (float*)walloc((size_t)BN * 64 * 4);
  float* VA = (float*)walloc((size_t)BN * 3 * 21 * 4);
  svblock(27, CS, 1440, 96, 64, CV, 478, 31, 21, BN, 64, 21, SA, 64, 0, VA, 21, 0);
  float* SB = (float*)walloc((size_t)BN * 64 * 4);
  float* VB = (float*)walloc((size_t)BN * 3 * 21 * 4);
  svblock(33, SA, 64, 0, 64, VA, 21, 0, 21, BN, 64, 21, SB, 64, 0, VB, 21, 0);
  float* sgp = (float*)walloc((size_t)B * 64 * 4);
  pool_N_s_k L1D((i64)B * 64)     (SB, 64, 0, B, N, 64, sgp, 0);
  float* vgp = (float*)walloc((size_t)B * 3 * 21 * 4);
  pool_N_v_k L1D((i64)B * 3 * 21) (VB, 21, 0, B, N, 21, vgp);
  float* ST = (float*)walloc((size_t)BN * 128 * 4);
  copy2d_k  L1D((i64)BN * 64) (CS, 1440, 96, BN, 64, ST, 128, 0);
  bcast_s_k L1D((i64)BN * 64) (sgp, B, N, 64, ST, 128, 64);
  float* VT = (float*)walloc((size_t)BN * 3 * 42 * 4);
  copy2d_k  L1D((i64)BN * 3 * 21) (CV, 478, 31, BN * 3, 21, VT, 42, 0);
  bcast_v_k L1D((i64)BN * 3 * 21) (vgp, B, N, 21, VT, 42, 21);

  // ---- c4, c5 ----
  svblock(39, ST, 128, 0, 128, VT, 42, 0, 42, BN, 256, 85,   CS, 1440, 160, CV, 478, 52);
  svblock(45, CS, 1440, 160, 256, CV, 478, 52, 85, BN, 1024, 341, CS, 1440, 416, CV, 478, 137);

  // ---- global mean pools, SF/VF ----
  float* smean = (float*)walloc((size_t)B * 1024 * 4);
  pool_N_s_k L1D((i64)B * 1024)    (CS, 1440, 416, B, N, 1024, smean, 1);
  float* vmean = (float*)walloc((size_t)B * 3 * 341 * 4);
  pool_N_v_k L1D((i64)B * 3 * 341) (CV, 478, 137, B, N, 341, vmean);
  float* XF = (float*)walloc((size_t)BN * 4094 * 4);        // [sf(2048) | vloc(2046)]
  copy2d_k  L1D((i64)BN * 1024) (CS, 1440, 416, BN, 1024, XF, 4094, 0);
  bcast_s_k L1D((i64)BN * 1024) (smean, B, N, 1024, XF, 4094, 1024);
  float* VF = (float*)walloc((size_t)BN * 3 * 682 * 4);
  copy2d_k  L1D((i64)BN * 3 * 341) (CV, 478, 137, BN * 3, 341, VF, 682, 0);
  bcast_v_k L1D((i64)BN * 3 * 341) (vmean, B, N, 341, VF, 682, 341);

  // ---- SVFuse: trans = VF @ fuse_Wt, vloc projection ----
  const int KpF = 704;
  bf16* VFb = (bf16*)walloc((size_t)BN * 3 * KpF * 2);
  pack_bf16_k L1D((i64)BN * 3 * KpF) (VF, 682, 0, BN * 3, 682, VFb, KpF);
  bf16* WtT = packW((const float*)d_in[51], 682, 3, KpF);
  float* TR = (float*)walloc((size_t)BN * 9 * 4);
  gemm(VFb, WtT, TR, BN * 3, 3, KpF, 3, 0);
  vloc_k L1D((i64)BN * 682) (TR, VF, BN, 682, XF, 4094, 2048);

  // ---- fu1 / fu2 / global max ----
  bf16* XFb = (bf16*)walloc((size_t)BN * 4096 * 2);
  pack_bf16_k L1D((i64)BN * 4096) (XF, 4094, 0, BN, 4094, XFb, 4096);
  bf16* H1b = (bf16*)walloc((size_t)BN * 512 * 2);
  lin_bn(XFb, BN, 4094, 4096, (const float*)d_in[52], (const float*)d_in[54],
         (const float*)d_in[55], 511, nullptr, 0, 0, H1b, 512);
  float* H2 = (float*)walloc((size_t)BN * 4094 * 4);
  lin_bn(H1b, BN, 511, 512, (const float*)d_in[56], (const float*)d_in[58],
         (const float*)d_in[59], 4094, H2, 4094, 0, nullptr, 0);
  float* G = (float*)walloc((size_t)B * 4094 * 4);
  pool_N_s_k L1D((i64)B * 4094) (H2, 4094, 0, B, N, 4094, G, 0);

  // ---- FEAT = [g(4094) | l(16) | cs(1440) | cvr(1434)] = 6984 ----
  float* FEAT = (float*)walloc((size_t)BN * 6984 * 4);
  bcast_s_k L1D((i64)BN * 4094) (G, B, N, 4094, FEAT, 6984, 0);
  bcast_s_k L1D((i64)BN * 16)   (lfe, B, N, 16, FEAT, 6984, 4094);
  copy2d_k  L1D((i64)BN * 1440) (CS, 1440, 0, BN, 1440, FEAT, 6984, 4110);
  cvr_k     L1D((i64)BN * 478)  (CV, 478, 0, TR, BN, 478, FEAT, 6984, 5550);

  // ---- s1..s4 head ----
  const int KpS1 = 7040;                    // pad64(6984)
  bf16* Fb = (bf16*)walloc((size_t)BN * KpS1 * 2);
  pack_bf16_k L1D((i64)BN * KpS1) (FEAT, 6984, 0, BN, 6984, Fb, KpS1);
  bf16* N1b = (bf16*)walloc((size_t)BN * 256 * 2);
  lin_bn(Fb, BN, 6984, KpS1, (const float*)d_in[60], (const float*)d_in[62],
         (const float*)d_in[63], 256, nullptr, 0, 0, N1b, 256);
  bf16* N2b = (bf16*)walloc((size_t)BN * 256 * 2);
  lin_bn(N1b, BN, 256, 256, (const float*)d_in[64], (const float*)d_in[66],
         (const float*)d_in[67], 256, nullptr, 0, 0, N2b, 256);
  bf16* N3b = (bf16*)walloc((size_t)BN * 128 * 2);
  lin_bn(N2b, BN, 256, 256, (const float*)d_in[68], (const float*)d_in[70],
         (const float*)d_in[71], 128, nullptr, 0, 0, N3b, 128);
  bf16* W4T = packW((const float*)d_in[72], 128, 50, 128);
  float* C50 = (float*)walloc((size_t)BN * 50 * 4);
  gemm(N3b, W4T, C50, BN, 50, 128, 50, 0);
  final_k L1D((i64)B * 50 * N) (C50, (const float*)d_in[73], B, N, (float*)d_out);
}
